// ResidualVectorQuantizer_20684562498311
// MI455X (gfx1250) — matrix operand order repaired and checked by reading, not compiled, not run on hardware
//
#include <hip/hip_runtime.h>

typedef __attribute__((ext_vector_type(16))) _Float16 v16h;
typedef __attribute__((ext_vector_type(8)))  float    v8f;
typedef __attribute__((ext_vector_type(4)))  unsigned int v4u;
typedef __attribute__((ext_vector_type(8)))  int      v8i;
typedef __attribute__((ext_vector_type(4)))  int      v4i;

#define N_ITEMS 262144
#define DIM     128
#define NCB     3
#define KCB     256
// packed codebook: halves per codebook = 256*128 = 32768 (64KB)
#define CB_HALVES 32768

// ---------------------------------------------------------------------------
// Pack codebooks (f32) into f16 WMMA B-fragment order:
// layout [cb][ntile(16)][kstep(4)][lane(32)][half(16)]
// B is K x N (32 x 16 per fragment). Per ISA 7.12.2 (dense 16-bit B):
//   lane L holds column n = L&15; halves h=0..15 hold K = k0 + (h&7) + 8*(L>=16) + 16*(h>=8)
// ---------------------------------------------------------------------------
__global__ void rvq_pack(const float* __restrict__ cbs, _Float16* __restrict__ bpack) {
    int gid  = blockIdx.x * 256 + threadIdx.x;     // 0 .. 98303
    int h    = gid & 15;
    int lane = (gid >> 4) & 31;
    int s    = (gid >> 9) & 3;
    int t    = (gid >> 11) & 15;
    int cb   = gid >> 15;
    int n    = t * 16 + (lane & 15);
    int k    = s * 32 + (h & 7) + ((lane >> 4) << 3) + ((h >> 3) << 4);
    bpack[gid] = (_Float16)cbs[(cb * KCB + n) * DIM + k];
}

// ||c||^2 per codeword (3*256 entries)
__global__ void rvq_cnorm(const float* __restrict__ cbs, float* __restrict__ cnorm) {
    int tid = blockIdx.x * 256 + threadIdx.x;
    if (tid >= NCB * KCB) return;
    const float* row = cbs + tid * DIM;
    float s = 0.f;
#pragma unroll
    for (int k = 0; k < DIM; ++k) s = fmaf(row[k], row[k], s);
    cnorm[tid] = s;
}

// load 16 floats in the A-fragment k-pattern: p[0..7] and p[16..23]
__device__ __forceinline__ void load16(const float* __restrict__ p, float v[16]) {
    float4 a0 = ((const float4*)p)[0];
    float4 a1 = ((const float4*)p)[1];
    float4 a2 = ((const float4*)(p + 16))[0];
    float4 a3 = ((const float4*)(p + 16))[1];
    v[0]=a0.x;  v[1]=a0.y;  v[2]=a0.z;  v[3]=a0.w;
    v[4]=a1.x;  v[5]=a1.y;  v[6]=a1.z;  v[7]=a1.w;
    v[8]=a2.x;  v[9]=a2.y;  v[10]=a2.z; v[11]=a2.w;
    v[12]=a3.x; v[13]=a3.y; v[14]=a3.z; v[15]=a3.w;
}

// ---------------------------------------------------------------------------
// Main kernel: 256 threads = 8 wave32; each wave owns 16 rows (block: 128 rows)
// ---------------------------------------------------------------------------
__launch_bounds__(256)
__global__ void rvq_main(const float* __restrict__ x,
                         const float* __restrict__ cbs,
                         const _Float16* __restrict__ bpack,
                         const float* __restrict__ cnorm,
                         float* __restrict__ out) {
    __shared__ __align__(32) _Float16 Bs[CB_HALVES];   // 64KB: current codebook, fragment order
    __shared__ float Cs[NCB * KCB];                    // 3KB: all ||c||^2
    __shared__ int idx_s[8][16];

    const int tid  = threadIdx.x;
    const int l    = tid & 31;          // lane in wave
    const int w    = tid >> 5;          // wave id in block
    const int half = l >> 4;            // 0: lanes 0-15, 1: lanes 16-31
    const int lp   = l & 15;
    const int row  = blockIdx.x * 128 + w * 16 + lp;   // A-fragment row of this lane
    const int aoff = half * 8;          // A k-offset: lane-half interleave of 8

    float* out_idx = out;                       // (n_items, 3) stored as float
    float* out_q   = out + N_ITEMS * NCB;       // (n_items, 128) f32

    // stage all ||c||^2 into LDS once (removes global load from inner loop)
#pragma unroll
    for (int i = tid; i < NCB * KCB; i += 256) Cs[i] = cnorm[i];

    // residual held in f32 registers, A-fragment k-order:
    // resid[s][0..7]  -> k = s*32 + aoff + 0..7
    // resid[s][8..15] -> k = s*32 + 16 + aoff + 0..7
    float resid[4][16];
#pragma unroll
    for (int s = 0; s < 4; ++s)
        load16(x + row * DIM + s * 32 + aoff, resid[s]);

    for (int c = 0; c < NCB; ++c) {
        // ---- stage packed codebook into LDS via Tensor Data Mover ----
        // 1-D tensor: 8192 elements of 8 bytes = 64KB, contiguous.
        if (w == 0) {
            const unsigned lds_base = (unsigned)(uintptr_t)(void*)Bs; // low 32b = LDS offset
            const unsigned long long ga =
                (unsigned long long)(uintptr_t)(bpack + (size_t)c * CB_HALVES);
            v4u g0;
            g0[0] = 1u;                                   // count=1, user descriptor
            g0[1] = lds_base;                             // lds_addr
            g0[2] = (unsigned)(ga & 0xffffffffu);         // global_addr[31:0]
            g0[3] = (unsigned)((ga >> 32) & 0x1ffffffu)   // global_addr[56:32]
                    | (2u << 30);                         // type = 2 (image)
            v8i g1;
            g1[0] = (int)(3u << 16);                      // data_size=3 (8B); wg_mask=0
            g1[1] = (int)((8192u & 0xffffu) << 16);       // tensor_dim0[15:0]
            g1[2] = 0;                                    // tensor_dim0[31:16], tensor_dim1 lo
            g1[3] = (int)(8192u << 16);                   // tile_dim0 = 8192
            g1[4] = 0;                                    // tile_dim1=0, tile_dim2=0 (1-D)
            g1[5] = 8192;                                 // tensor_dim0_stride[31:0]
            g1[6] = 0;
            g1[7] = 0;
            v4i z4 = {0, 0, 0, 0};
            v8i z8 = {0, 0, 0, 0, 0, 0, 0, 0};
            // amdgpu-toolchain (clang-23) 6-arg form:
            // (group0, group1, group2, group3, <extra>, cpol)
            __builtin_amdgcn_tensor_load_to_lds(g0, g1, z4, z4, z8, 0);
            __builtin_amdgcn_s_wait_tensorcnt(0);
        }
        __syncthreads();

        // ---- convert residual to f16 A fragments ----
        v16h af[4];
#pragma unroll
        for (int s = 0; s < 4; ++s)
#pragma unroll
            for (int h = 0; h < 16; ++h)
                af[s][h] = (_Float16)resid[s][h];

        float bestv[8];
        int   bestn[8];
#pragma unroll
        for (int j = 0; j < 8; ++j) { bestv[j] = 3.0e38f; bestn[j] = 0; }

        // ---- score GEMM: 16 N-tiles x 4 K-steps of v_wmma_f32_16x16x32_f16 ----
        for (int t = 0; t < 16; ++t) {
            v8f acc;
#pragma unroll
            for (int j = 0; j < 8; ++j) acc[j] = 0.f;
#pragma unroll
            for (int s = 0; s < 4; ++s) {
                v16h bf = *((const v16h*)(Bs + ((t * 4 + s) * 32 + l) * 16));
                acc = __builtin_amdgcn_wmma_f32_16x16x32_f16(
                          false, af[s], false, bf, (short)0, acc, false, false);
            }
            // lane's column n; C layout: slot j -> row j + 8*half, col lp
            const int   n  = t * 16 + lp;
            const float cn = Cs[c * KCB + n];
#pragma unroll
            for (int j = 0; j < 8; ++j) {
                float v = fmaf(-2.0f, acc[j], cn);   // ||c||^2 - 2 r.c  (drop ||r||^2)
                if (v < bestv[j]) { bestv[j] = v; bestn[j] = n; }
            }
        }

        // ---- argmin across the 16 lanes of each half (ties -> lower index) ----
#pragma unroll
        for (int m = 1; m <= 8; m <<= 1) {
#pragma unroll
            for (int j = 0; j < 8; ++j) {
                float ov = __shfl_xor(bestv[j], m, 32);
                int   on = __shfl_xor(bestn[j], m, 32);
                if (ov < bestv[j] || (ov == bestv[j] && on < bestn[j])) {
                    bestv[j] = ov; bestn[j] = on;
                }
            }
        }
        // slot j of half 0 -> row j; half 1 -> row j+8
        if (lp == 0) {
#pragma unroll
            for (int j = 0; j < 8; ++j) idx_s[w][half * 8 + j] = bestn[j];
        }
        __syncthreads();

        const int myidx = idx_s[w][lp];
        if (half == 0) out_idx[row * NCB + c] = (float)myidx;

        // ---- gather f32 codeword, update f32 residual ----
        const float* cq = cbs + (c * KCB + myidx) * DIM;
#pragma unroll
        for (int s = 0; s < 4; ++s) {
            float q[16];
            load16(cq + s * 32 + aoff, q);
#pragma unroll
            for (int h = 0; h < 16; ++h) resid[s][h] -= q[h];
        }
    }

    // ---- quantized = x - final residual ----
#pragma unroll
    for (int s = 0; s < 4; ++s) {
        float xv[16];
        const float* px = x + row * DIM + s * 32 + aoff;
        load16(px, xv);
        float* po = out_q + row * DIM + s * 32 + aoff;
        float4 o0, o1, o2, o3;
        o0.x = xv[0]  - resid[s][0];  o0.y = xv[1]  - resid[s][1];
        o0.z = xv[2]  - resid[s][2];  o0.w = xv[3]  - resid[s][3];
        o1.x = xv[4]  - resid[s][4];  o1.y = xv[5]  - resid[s][5];
        o1.z = xv[6]  - resid[s][6];  o1.w = xv[7]  - resid[s][7];
        o2.x = xv[8]  - resid[s][8];  o2.y = xv[9]  - resid[s][9];
        o2.z = xv[10] - resid[s][10]; o2.w = xv[11] - resid[s][11];
        o3.x = xv[12] - resid[s][12]; o3.y = xv[13] - resid[s][13];
        o3.z = xv[14] - resid[s][14]; o3.w = xv[15] - resid[s][15];
        ((float4*)po)[0] = o0;
        ((float4*)po)[1] = o1;
        ((float4*)(po + 16))[0] = o2;
        ((float4*)(po + 16))[1] = o3;
    }
}

extern "C" void kernel_launch(void* const* d_in, const int* in_sizes, int n_in,
                              void* d_out, int out_size, void* d_ws, size_t ws_size,
                              hipStream_t stream) {
    const float* x   = (const float*)d_in[0];
    const float* cbs = (const float*)d_in[1];

    _Float16* bpack = (_Float16*)d_ws;
    float* cnorm = (float*)((char*)d_ws + (size_t)NCB * CB_HALVES * sizeof(_Float16));

    rvq_pack<<<(NCB * CB_HALVES) / 256, 256, 0, stream>>>(cbs, bpack);
    rvq_cnorm<<<3, 256, 0, stream>>>(cbs, cnorm);
    rvq_main<<<N_ITEMS / 128, 256, 0, stream>>>(x, cbs, bpack, cnorm, (float*)d_out);
}
